// MPNEncoder_84250078478375
// MI455X (gfx1250) — matrix operand order repaired and checked by reading, not compiled, not run on hardware
//
#include <hip/hip_runtime.h>
#include <stdint.h>

// ---------------------------------------------------------------------------
// MPNEncoder for MI455X (gfx1250): all GEMMs via V_WMMA_F32_16X16X32_F16,
// operand staging via GLOBAL_LOAD_ASYNC_TO_LDS_B128 + double-buffered LDS.
// ---------------------------------------------------------------------------

typedef __attribute__((ext_vector_type(16))) _Float16 v16h;
typedef __attribute__((ext_vector_type(8)))  _Float16 v8h;
typedef __attribute__((ext_vector_type(8)))  float    v8f;

#define HID    512
#define NMOLS  256
#define SEQ    64
#define MAXNB  6
#define NATOMS (1 + NMOLS * SEQ)          // 16385
#define NBONDS (1 + NMOLS * SEQ * 4)      // 65537
#define KFPAD  160                        // 133/147 padded to multiple of 32
#define NDEPTH 5                          // DEPTH-1 W_h layers

// ----------------------------- WMMA GEMM -----------------------------------
// C[M,N] = epi( A[M,K](f16,row,lda,+abase) @ W[N,K](f16,row,ld=K)^T )
// epilogue: +bias[N] (opt), +addsrc[M,N] (opt), relu (opt). C row stride = N.
#define BM 128
#define BN 64
#define BK 32

__device__ __forceinline__ unsigned lds_off32(const void* p)
{
    // generic pointer to __shared__: low 32 bits == LDS byte offset
    return (unsigned)(uintptr_t)p;
}

__device__ __forceinline__ void async_b128(unsigned lds, const void* gaddr)
{
    asm volatile("global_load_async_to_lds_b128 %0, %1, off"
                 :: "v"(lds), "v"(gaddr) : "memory");
}

__global__ __launch_bounds__(256)
void gemm16_wmma(const _Float16* __restrict__ A, int lda, long abase,
                 const _Float16* __restrict__ W,
                 float* __restrict__ C,
                 int M, int N, int K,
                 const float* __restrict__ bias,
                 const float* __restrict__ addsrc,
                 int relu_flag)
{
    __shared__ _Float16 As[2][BM][BK + 8];   // +8 halves pad: row = 80B, 16B aligned
    __shared__ _Float16 Bs[2][BN][BK + 8];

    const int tid  = threadIdx.x;
    const int wave = tid >> 5;
    const int lane = tid & 31;
    const int m0   = blockIdx.y * BM;
    const int n0   = blockIdx.x * BN;

    const int l   = lane & 15;               // row-within-16 for fragments
    const int hiL = lane >> 4;               // 0 = lanes 0-15, 1 = lanes 16-31
    const int wm  = wave * 16;               // wave's M sub-tile

    // staging coordinates: A tile 128x32 halves -> 2 x b128 per thread (same row)
    const int ar = tid >> 1;                 // 0..127
    const int ac = (tid & 1) * 16;           // 0 or 16 (two b128 chunks: ac, ac+8)
    const int br = tid >> 2;                 // 0..63
    const int bc = (tid & 3) * 8;            // 0,8,16,24

    // clamp OOB rows (their C rows are masked in the epilogue; reads stay in-buffer)
    int arow = m0 + ar; if (arow >= M) arow = M - 1;
    const _Float16* gA = A + abase + (size_t)arow * lda + ac;
    const _Float16* gB = W + (size_t)(n0 + br) * K + bc;

    v8f acc[4] = {v8f{}, v8f{}, v8f{}, v8f{}};

    // ---- prologue: async-stage first K tile into buffer 0 ----
    async_b128(lds_off32(&As[0][ar][ac]),     gA);
    async_b128(lds_off32(&As[0][ar][ac + 8]), gA + 8);
    async_b128(lds_off32(&Bs[0][br][bc]),     gB);

    for (int kt = 0; kt < K; kt += BK) {
        const int par = (kt >> 5) & 1;
        if (kt + BK < K) {
            // issue next tile into the other buffer, overlap with this tile's WMMAs
            async_b128(lds_off32(&As[par ^ 1][ar][ac]),     gA + (kt + BK));
            async_b128(lds_off32(&As[par ^ 1][ar][ac + 8]), gA + (kt + BK) + 8);
            async_b128(lds_off32(&Bs[par ^ 1][br][bc]),     gB + (kt + BK));
            asm volatile("s_wait_asynccnt 0x3" ::: "memory");  // current tile landed
        } else {
            asm volatile("s_wait_asynccnt 0x0" ::: "memory");
        }
        __syncthreads();                      // all waves' async writes visible

        // A fragment (16x32 f16): M=l; K = h + (h>=8?8:0) + hiL*8
        v8h alo = *(const v8h*)&As[par][wm + l][hiL * 8];
        v8h ahi = *(const v8h*)&As[par][wm + l][hiL * 8 + 16];
        v16h afrag = __builtin_shufflevector(alo, ahi,
                        0,1,2,3,4,5,6,7,8,9,10,11,12,13,14,15);

#pragma unroll
        for (int nt = 0; nt < 4; ++nt) {
            // B fragment (32x16 f16): N=l; K = h + (h>=8?8:0) + hiL*8
            v8h blo = *(const v8h*)&Bs[par][nt * 16 + l][hiL * 8];
            v8h bhi = *(const v8h*)&Bs[par][nt * 16 + l][hiL * 8 + 16];
            v16h bfrag = __builtin_shufflevector(blo, bhi,
                            0,1,2,3,4,5,6,7,8,9,10,11,12,13,14,15);
            acc[nt] = __builtin_amdgcn_wmma_f32_16x16x32_f16(
                          false, afrag, false, bfrag,
                          (short)0, acc[nt], false, false);
        }
        __syncthreads();                      // reads done before buffer is re-filled
    }

    // ---- epilogue: C/D layout — VGPR r: lanes 0-15 M=r, lanes 16-31 M=r+8
#pragma unroll
    for (int nt = 0; nt < 4; ++nt) {
        int col = n0 + nt * 16 + l;
        float bv = bias ? bias[col] : 0.f;
#pragma unroll
        for (int r = 0; r < 8; ++r) {
            int row = m0 + wm + r + hiL * 8;
            if (row < M) {
                float v = acc[nt][r] + bv;
                if (addsrc)    v += addsrc[(size_t)row * N + col];
                if (relu_flag) v  = fmaxf(v, 0.f);
                C[(size_t)row * N + col] = v;
            }
        }
    }
}

// ------------------------- elementwise kernels ------------------------------

// f32 -> f16 with K padding (src rows x ks -> dst rows x kd, zero pad)
__global__ void cvt_pad_f16(const float* __restrict__ src, _Float16* __restrict__ dst,
                            int rows, int ks, int kd)
{
    int i = blockIdx.x * blockDim.x + threadIdx.x;
    if (i >= rows * kd) return;
    int r = i / kd, c = i % kd;
    dst[i] = (c < ks) ? (_Float16)src[(size_t)r * ks + c] : (_Float16)0.f;
}

// message_atom += sum(nei) * max(nei)   (nei = mb[a2b], zeroed where idx==0)
__global__ void agg_update(const float* __restrict__ mb, const int* __restrict__ a2b,
                           float* __restrict__ message_atom)
{
    int i = blockIdx.x * blockDim.x + threadIdx.x;
    if (i >= NATOMS * HID) return;
    int a = i >> 9, h = i & (HID - 1);
    float s = 0.f, m = -1e30f;
#pragma unroll
    for (int nb = 0; nb < MAXNB; ++nb) {
        int idx = a2b[a * MAXNB + nb];
        float v = (idx == 0) ? 0.f : mb[(size_t)idx * HID + h];
        s += v;
        m = fmaxf(m, v);
    }
    message_atom[i] += s * m;
}

// tmp16[b] = f16( message_atom[b2a[b]] - mb[b2revb[b]] )
__global__ void bond_diff(const float* __restrict__ message_atom,
                          const float* __restrict__ mb,
                          const int* __restrict__ b2a,
                          const int* __restrict__ b2revb,
                          _Float16* __restrict__ tmp16)
{
    int i = blockIdx.x * blockDim.x + threadIdx.x;
    if (i >= NBONDS * HID) return;
    int b = i >> 9, h = i & (HID - 1);
    float v = message_atom[(size_t)b2a[b] * HID + h]
            - mb[(size_t)b2revb[b] * HID + h];
    tmp16[i] = (_Float16)v;
}

// concat16[a] = [ sum*max agg | message_atom[a] | input_atom[a] ]  (f16, A x 1536)
__global__ void final_concat(const float* __restrict__ mb, const int* __restrict__ a2b,
                             const float* __restrict__ message_atom,
                             const float* __restrict__ input_atom,
                             _Float16* __restrict__ concat16)
{
    int i = blockIdx.x * blockDim.x + threadIdx.x;
    if (i >= NATOMS * HID) return;
    int a = i >> 9, h = i & (HID - 1);
    float s = 0.f, m = -1e30f;
#pragma unroll
    for (int nb = 0; nb < MAXNB; ++nb) {
        int idx = a2b[a * MAXNB + nb];
        float v = (idx == 0) ? 0.f : mb[(size_t)idx * HID + h];
        s += v;
        m = fmaxf(m, v);
    }
    size_t base = (size_t)a * (3 * HID);
    concat16[base + h]            = (_Float16)(s * m);
    concat16[base + HID + h]      = (_Float16)message_atom[i];
    concat16[base + 2 * HID + h]  = (_Float16)input_atom[i];
}

// msg = relu(node + gru_bias) -> msg16; also message16 row0 = [msg0 | msg0]
__global__ void msg_relu(const float* __restrict__ node, const float* __restrict__ gru_bias,
                         _Float16* __restrict__ msg16, _Float16* __restrict__ message16)
{
    int i = blockIdx.x * blockDim.x + threadIdx.x;
    if (i >= NATOMS * HID) return;
    int a = i >> 9, h = i & (HID - 1);
    float v = fmaxf(node[i] + gru_bias[h], 0.f);
    msg16[i] = (_Float16)v;
    if (a == 0) {
        message16[h]       = (_Float16)v;
        message16[HID + h] = (_Float16)v;
    }
}

// h0[m] = max_t node[1 + m*SEQ + t]   -> hbuf (f32) + h16 (f16)
__global__ void h0_max(const float* __restrict__ node,
                       float* __restrict__ hbuf, _Float16* __restrict__ h16)
{
    int i = blockIdx.x * blockDim.x + threadIdx.x;
    if (i >= NMOLS * HID) return;
    int m = i >> 9, h = i & (HID - 1);
    float mx = -1e30f;
    for (int t = 0; t < SEQ; ++t)
        mx = fmaxf(mx, node[(size_t)(1 + m * SEQ + t) * HID + h]);
    hbuf[i] = mx;
    h16[i]  = (_Float16)mx;
}

// one GRU step: gates from gi (x@Wih^T+bih) and gh (h@Whh^T+bhh)
__global__ void gru_step(const float* __restrict__ gi, const float* __restrict__ gh,
                         float* __restrict__ hbuf, _Float16* __restrict__ h16,
                         _Float16* __restrict__ message16, int tpos, int coloff)
{
    int i = blockIdx.x * blockDim.x + threadIdx.x;
    if (i >= NMOLS * HID) return;
    int m = i >> 9, h = i & (HID - 1);
    size_t g = (size_t)m * (3 * HID);
    float ir = gi[g + h],            hr = gh[g + h];
    float iz = gi[g + HID + h],      hz = gh[g + HID + h];
    float in = gi[g + 2 * HID + h],  hn = gh[g + 2 * HID + h];
    float r = 1.f / (1.f + __expf(-(ir + hr)));
    float z = 1.f / (1.f + __expf(-(iz + hz)));
    float n = tanhf(in + r * hn);
    float hnew = (1.f - z) * n + z * hbuf[i];
    hbuf[i] = hnew;
    h16[i]  = (_Float16)hnew;
    message16[(size_t)(1 + m * SEQ + tpos) * (2 * HID) + coloff + h] = (_Float16)hnew;
}

// ------------------------------ launcher ------------------------------------

extern "C" void kernel_launch(void* const* d_in, const int* in_sizes, int n_in,
                              void* d_out, int out_size, void* d_ws, size_t ws_size,
                              hipStream_t stream)
{
    const float* f_atoms   = (const float*)d_in[0];
    const float* f_bonds   = (const float*)d_in[1];
    const int*   a2b       = (const int*)  d_in[2];
    const int*   b2a       = (const int*)  d_in[3];
    const int*   b2revb    = (const int*)  d_in[4];
    const float* W_i_atom  = (const float*)d_in[5];
    const float* W_i_bond  = (const float*)d_in[6];
    const float* W_h       = (const float*)d_in[7];
    const float* W_lr      = (const float*)d_in[8];
    const float* W_o       = (const float*)d_in[9];
    const float* b_o       = (const float*)d_in[10];
    const float* gru_bias  = (const float*)d_in[11];
    const float* gw_ih[2]  = {(const float*)d_in[12], (const float*)d_in[16]};
    const float* gw_hh[2]  = {(const float*)d_in[13], (const float*)d_in[17]};
    const float* gb_ih[2]  = {(const float*)d_in[14], (const float*)d_in[18]};
    const float* gb_hh[2]  = {(const float*)d_in[15], (const float*)d_in[19]};

    float* out = (float*)d_out;

    // ---- carve workspace (256B aligned) ----
    char* p = (char*)d_ws;
    auto carve = [&](size_t bytes) -> void* {
        void* r = (void*)p;
        p += (bytes + 255) & ~(size_t)255;
        return r;
    };
    float* input_atom   = (float*)carve((size_t)NATOMS * HID * 4);
    float* input_bond   = (float*)carve((size_t)NBONDS * HID * 4);
    float* message_atom = (float*)carve((size_t)NATOMS * HID * 4);
    float* mb           = (float*)carve((size_t)NBONDS * HID * 4);
    float* node         = (float*)carve((size_t)NATOMS * HID * 4);
    float* gi           = (float*)carve((size_t)NMOLS * 3 * HID * 4);
    float* gh           = (float*)carve((size_t)NMOLS * 3 * HID * 4);
    float* hbuf         = (float*)carve((size_t)NMOLS * HID * 4);

    _Float16* fa16      = (_Float16*)carve((size_t)NATOMS * KFPAD * 2);
    _Float16* fb16      = (_Float16*)carve((size_t)NBONDS * KFPAD * 2);
    _Float16* tmp16     = (_Float16*)carve((size_t)NBONDS * HID * 2);
    _Float16* concat16  = (_Float16*)carve((size_t)NATOMS * 3 * HID * 2);
    _Float16* msg16     = (_Float16*)carve((size_t)NATOMS * HID * 2);
    _Float16* h16       = (_Float16*)carve((size_t)NMOLS * HID * 2);
    _Float16* message16 = (_Float16*)carve((size_t)NATOMS * 2 * HID * 2);
    _Float16* WiA16     = (_Float16*)carve((size_t)HID * KFPAD * 2);
    _Float16* WiB16     = (_Float16*)carve((size_t)HID * KFPAD * 2);
    _Float16* Wh16      = (_Float16*)carve((size_t)NDEPTH * HID * HID * 2);
    _Float16* Wlr16     = (_Float16*)carve((size_t)HID * 3 * HID * 2);
    _Float16* Wo16      = (_Float16*)carve((size_t)HID * 2 * HID * 2);
    _Float16* gwih16[2] = {(_Float16*)carve((size_t)3 * HID * HID * 2),
                           (_Float16*)carve((size_t)3 * HID * HID * 2)};
    _Float16* gwhh16[2] = {(_Float16*)carve((size_t)3 * HID * HID * 2),
                           (_Float16*)carve((size_t)3 * HID * HID * 2)};

    auto ew = [&](int total) { return dim3((total + 255) / 256); };
    auto cvt = [&](const float* s, _Float16* d, int rows, int ks, int kd) {
        cvt_pad_f16<<<ew(rows * kd), 256, 0, stream>>>(s, d, rows, ks, kd);
    };
    auto gemm = [&](const _Float16* A, int lda, long abase, const _Float16* W,
                    float* C, int M, int N, int K,
                    const float* bias, const float* addsrc, int relu) {
        dim3 grid(N / BN, (M + BM - 1) / BM);
        gemm16_wmma<<<grid, 256, 0, stream>>>(A, lda, abase, W, C, M, N, K,
                                              bias, addsrc, relu);
    };

    // ---- f16 conversions (weights + features, K padded) ----
    cvt(f_atoms, fa16, NATOMS, 133, KFPAD);
    cvt(f_bonds, fb16, NBONDS, 147, KFPAD);
    cvt(W_i_atom, WiA16, HID, 133, KFPAD);
    cvt(W_i_bond, WiB16, HID, 147, KFPAD);
    cvt(W_h, Wh16, NDEPTH * HID, HID, HID);
    cvt(W_lr, Wlr16, HID, 3 * HID, 3 * HID);
    cvt(W_o, Wo16, HID, 2 * HID, 2 * HID);
    for (int d = 0; d < 2; ++d) {
        cvt(gw_ih[d], gwih16[d], 3 * HID, HID, HID);
        cvt(gw_hh[d], gwhh16[d], 3 * HID, HID, HID);
    }

    // ---- input layers ----
    gemm(fa16, KFPAD, 0, WiA16, input_atom, NATOMS, HID, KFPAD, nullptr, nullptr, 1);
    gemm(fb16, KFPAD, 0, WiB16, input_bond, NBONDS, HID, KFPAD, nullptr, nullptr, 1);
    hipMemcpyAsync(message_atom, input_atom, (size_t)NATOMS * HID * 4,
                   hipMemcpyDeviceToDevice, stream);
    hipMemcpyAsync(mb, input_bond, (size_t)NBONDS * HID * 4,
                   hipMemcpyDeviceToDevice, stream);

    // ---- message passing depth loop ----
    for (int d = 0; d < NDEPTH; ++d) {
        agg_update<<<ew(NATOMS * HID), 256, 0, stream>>>(mb, a2b, message_atom);
        bond_diff<<<ew(NBONDS * HID), 256, 0, stream>>>(message_atom, mb, b2a, b2revb, tmp16);
        gemm(tmp16, HID, 0, Wh16 + (size_t)d * HID * HID, mb,
             NBONDS, HID, HID, nullptr, input_bond, 1);
    }

    // ---- readout: concat + W_lr ----
    final_concat<<<ew(NATOMS * HID), 256, 0, stream>>>(mb, a2b, message_atom,
                                                       input_atom, concat16);
    gemm(concat16, 3 * HID, 0, Wlr16, node, NATOMS, HID, 3 * HID, nullptr, nullptr, 0);
    msg_relu<<<ew(NATOMS * HID), 256, 0, stream>>>(node, gru_bias, msg16, message16);

    // ---- bidirectional BatchGRU ----
    for (int dir = 0; dir < 2; ++dir) {
        h0_max<<<ew(NMOLS * HID), 256, 0, stream>>>(node, hbuf, h16);
        for (int s = 0; s < SEQ; ++s) {
            int t = dir ? (SEQ - 1 - s) : s;
            // gi = x_t @ w_ih^T + b_ih   (x_t rows strided through msg16)
            gemm(msg16, SEQ * HID, (long)(1 + t) * HID, gwih16[dir], gi,
                 NMOLS, 3 * HID, HID, gb_ih[dir], nullptr, 0);
            // gh = h @ w_hh^T + b_hh
            gemm(h16, HID, 0, gwhh16[dir], gh,
                 NMOLS, 3 * HID, HID, gb_hh[dir], nullptr, 0);
            gru_step<<<ew(NMOLS * HID), 256, 0, stream>>>(gi, gh, hbuf, h16,
                                                          message16, t, dir * HID);
        }
    }

    // ---- output layer: relu(message @ W_o^T + b_o) ----
    gemm(message16, 2 * HID, 0, Wo16, out, NATOMS, HID, 2 * HID, b_o, nullptr, 1);

    (void)in_sizes; (void)n_in; (void)out_size; (void)ws_size;
}
